// TopKTokenChoiceRouter_10402410791239
// MI455X (gfx1250) — matrix operand order repaired and checked by reading, not compiled, not run on hardware
//
#include <hip/hip_runtime.h>
#include <hip/hip_bf16.h>
#include <math.h>

// Problem constants (match reference)
#define SLEN 2048
#define BSZ  4
#define HDIM 4096
#define NEXP 64
#define NTOK (SLEN * BSZ)     // 8192 tokens
#define TOPK 2
#define Z_COEFF   1e-3f
#define AUX_COEFF 1e-2f

// Tiling
#define TILE_M  128           // tokens per block
#define KC      64            // K-chunk: two 16x16x32 k-steps per barrier
#define LSTRIDE 68            // padded logits row stride (floats), 16B-aligned, kills bank conflicts

typedef __attribute__((ext_vector_type(16))) __bf16 v16bf;
typedef __attribute__((ext_vector_type(8)))  float  v8f;
typedef __attribute__((ext_vector_type(4)))  float  vf4;

static __device__ __forceinline__ unsigned short f2bf(float f) {
  unsigned int u = __float_as_uint(f);
  u += 0x7FFFu + ((u >> 16) & 1u);   // round-to-nearest-even
  return (unsigned short)(u >> 16);
}

// Stage one K-chunk: A = x[tok0..tok0+127][k0..k0+63] (NT loads, streamed once),
//                    B = W[0..63][k0..k0+63]           (RT loads, L2-resident).
static __device__ __forceinline__ void stage_tiles(
    const float* __restrict__ x, const float* __restrict__ W,
    unsigned short* __restrict__ Ab, unsigned short* __restrict__ Bb,
    int tok0, int k0, int tid) {
#pragma unroll
  for (int i = 0; i < 8; ++i) {                // 2048 float4 / 256 threads
    int idx = tid + i * 256;
    int r = idx >> 4, c4 = (idx & 15) << 2;
    vf4 v = __builtin_nontemporal_load(
        (const vf4*)(x + (size_t)(tok0 + r) * HDIM + k0 + c4));
    unsigned int lo = ((unsigned int)f2bf(v.y) << 16) | f2bf(v.x);
    unsigned int hi = ((unsigned int)f2bf(v.w) << 16) | f2bf(v.z);
    *(uint2*)(&Ab[r * KC + c4]) = make_uint2(lo, hi);
  }
#pragma unroll
  for (int i = 0; i < 4; ++i) {                // 1024 float4 / 256 threads
    int idx = tid + i * 256;
    int r = idx >> 4, c4 = (idx & 15) << 2;
    vf4 v = *(const vf4*)(W + (size_t)r * HDIM + k0 + c4);
    unsigned int lo = ((unsigned int)f2bf(v.y) << 16) | f2bf(v.x);
    unsigned int hi = ((unsigned int)f2bf(v.w) << 16) | f2bf(v.z);
    *(uint2*)(&Bb[r * KC + c4]) = make_uint2(lo, hi);
  }
}

__global__ __launch_bounds__(256)
void router_gemm_topk(const float* __restrict__ x, const float* __restrict__ W,
                      float* __restrict__ out,
                      float* __restrict__ g_agg, float* __restrict__ g_cnt,
                      float* __restrict__ g_z) {
  __shared__ unsigned short Abuf[2][TILE_M * KC];   // 2 x 16 KB (ping-pong)
  __shared__ unsigned short Bbuf[2][NEXP * KC];     // 2 x  8 KB (ping-pong)
  __shared__ float          Lg[TILE_M * LSTRIDE];   // ~34 KB (logits, then probs)
  __shared__ float          scnt[NEXP];
  __shared__ float          zacc;

  const int tid   = threadIdx.x;
  const int wave  = tid >> 5;        // 0..7 -> M-tile
  const int lane  = tid & 31;
  const int lrow  = lane & 15;
  const int khalf = lane >> 4;       // 0 or 1
  const int tok0  = blockIdx.x * TILE_M;

  v8f acc[4] = {};                   // 4 N-tiles of 16 experts each

  stage_tiles(x, W, Abuf[0], Bbuf[0], tok0, 0, tid);
  __syncthreads();

  int buf = 0;
  for (int k0 = 0; k0 < HDIM; k0 += KC, buf ^= 1) {
    // ---- prefetch + stage next chunk into the other buffer ----
    if (k0 + KC < HDIM) {
      stage_tiles(x, W, Abuf[buf ^ 1], Bbuf[buf ^ 1], tok0, k0 + KC, tid);
      if (k0 + 2 * KC < HDIM) {      // global_prefetch_b8 for chunk after next
        __builtin_prefetch(
            x + (size_t)(tok0 + (tid >> 1)) * HDIM + k0 + 2 * KC + (tid & 1) * 32, 0, 0);
      }
    }

    // ---- compute current chunk: 2 k-steps x 4 expert tiles = 8 WMMAs/wave ----
    const unsigned short* Ab = Abuf[buf];
    const unsigned short* Bb = Bbuf[buf];
#pragma unroll
    for (int s = 0; s < 2; ++s) {
      const unsigned short* arow = &Ab[(wave * 16 + lrow) * KC + s * 32 + khalf * 8];
      union { uint4 u[2]; v16bf v; } af;
      af.u[0] = *(const uint4*)(arow);
      af.u[1] = *(const uint4*)(arow + 16);
#pragma unroll
      for (int nt = 0; nt < 4; ++nt) {
        const unsigned short* brow = &Bb[(nt * 16 + lrow) * KC + s * 32 + khalf * 8];
        union { uint4 u[2]; v16bf v; } bf;
        bf.u[0] = *(const uint4*)(brow);
        bf.u[1] = *(const uint4*)(brow + 16);
        acc[nt] = __builtin_amdgcn_wmma_f32_16x16x32_bf16(
            false, af.v, false, bf.v, (short)0, acc[nt], false, false);
      }
    }
    __syncthreads();
  }

  // ---- spill logits tile to padded LDS; zero reduction scratch ----
  if (tid < NEXP) scnt[tid] = 0.0f;
  if (tid == 0)   zacc = 0.0f;
#pragma unroll
  for (int nt = 0; nt < 4; ++nt) {
#pragma unroll
    for (int r = 0; r < 8; ++r) {
      int m = wave * 16 + r + khalf * 8;   // C layout: vgpr r = row r (lanes 0-15) / r+8 (16-31)
      int n = nt * 16 + lrow;
      Lg[m * LSTRIDE + n] = acc[nt][r];
    }
  }
  __syncthreads();

  // ---- per-token softmax, top-2, z-loss partial ----
  if (tid < TILE_M) {
    float* row = &Lg[tid * LSTRIDE];
    float l[NEXP];
    float mx = -INFINITY;
#pragma unroll
    for (int e = 0; e < NEXP; ++e) { l[e] = row[e]; mx = fmaxf(mx, l[e]); }
    float se = 0.0f;
#pragma unroll
    for (int e = 0; e < NEXP; ++e) { float p = __expf(l[e] - mx); l[e] = p; se += p; }
    float inv = 1.0f / se;
    float lse = mx + __logf(se);
    atomicAdd(&zacc, lse * lse);

    float b1 = -1.0f, b2 = -1.0f; int i1 = 0, i2 = 0;
#pragma unroll
    for (int e = 0; e < NEXP; ++e) {
      float p = l[e] * inv;
      row[e] = p;                       // keep probs for column sums
      if (p > b1)      { b2 = b1; i2 = i1; b1 = p; i1 = e; }
      else if (p > b2) { b2 = p; i2 = e; }
    }
    int t = tok0 + tid;
    out[t * TOPK + 0] = b1;
    out[t * TOPK + 1] = b2;
    out[NTOK * TOPK + t * TOPK + 0] = (float)i1;
    out[NTOK * TOPK + t * TOPK + 1] = (float)i2;
    atomicAdd(&scnt[i1], 1.0f);
    atomicAdd(&scnt[i2], 1.0f);
  }
  __syncthreads();

  // ---- per-expert block reductions -> one global atomic per expert ----
  if (tid < NEXP) {
    float s = 0.0f;
    for (int r = 0; r < TILE_M; ++r) s += Lg[r * LSTRIDE + tid];
    atomicAdd(&g_agg[tid], s);
    float c = scnt[tid];
    if (c != 0.0f) atomicAdd(&g_cnt[tid], c);
  }
  if (tid == 0) atomicAdd(g_z, zacc);
}

__global__ void router_finalize(const float* __restrict__ g_agg,
                                const float* __restrict__ g_cnt,
                                const float* __restrict__ g_z,
                                float* __restrict__ out) {
  __shared__ float part[NEXP];
  int t = threadIdx.x;
  part[t] = g_agg[t] * g_cnt[t];
  __syncthreads();
  if (t == 0) {
    float s = 0.0f;
    for (int e = 0; e < NEXP; ++e) s += part[e];
    const float lbl_scale =
        (float)NEXP * AUX_COEFF / ((float)NTOK * (float)NTOK * (float)TOPK);
    out[2 * NTOK * TOPK + 0] = g_z[0] * (Z_COEFF / (float)NTOK);  // z_loss
    out[2 * NTOK * TOPK + 1] = s * lbl_scale;                     // lbl_loss
  }
}

extern "C" void kernel_launch(void* const* d_in, const int* in_sizes, int n_in,
                              void* d_out, int out_size, void* d_ws, size_t ws_size,
                              hipStream_t stream) {
  const float* x = (const float*)d_in[0];   // [SL, BS, H] f32
  const float* W = (const float*)d_in[1];   // [E, H] f32
  float* out = (float*)d_out;

  float* g_agg = (float*)d_ws;              // [64]
  float* g_cnt = g_agg + NEXP;              // [64]
  float* g_z   = g_cnt + NEXP;              // [1]

  hipMemsetAsync(d_ws, 0, (2 * NEXP + 1) * sizeof(float), stream);
  router_gemm_topk<<<NTOK / TILE_M, 256, 0, stream>>>(x, W, out, g_agg, g_cnt, g_z);
  router_finalize<<<1, NEXP, 0, stream>>>(g_agg, g_cnt, g_z, out);
}
